// DecoderBlockWithRoPE_5351529251394
// MI455X (gfx1250) — matrix operand order repaired and checked by reading, not compiled, not run on hardware
//
#include <hip/hip_runtime.h>
#include <math.h>

// ---------------- constants (match reference) ----------------
#define BB   2
#define SS   2048
#define HH   2048
#define NHH  16
#define DD   128          // head dim
#define FFF  8192
#define MM   (BB*SS)      // 4096 rows
#define SOFTMAX_SCALE 0.08838834764831845f  // 1/sqrt(128)

typedef __attribute__((ext_vector_type(8)))  __bf16 v8bf;
typedef __attribute__((ext_vector_type(16))) __bf16 v16bf;
typedef __attribute__((ext_vector_type(8)))  float  v8f;

static __device__ inline v16bf make16(v8bf lo, v8bf hi) {
    v16bf r;
#pragma unroll
    for (int i = 0; i < 8; ++i) { r[i] = lo[i]; r[i + 8] = hi[i]; }
    return r;
}

static __device__ inline v8f vzero8() {
    v8f z = {0.f,0.f,0.f,0.f,0.f,0.f,0.f,0.f};
    return z;
}

static __device__ inline float gelu_tanh(float x) {
    const float k0 = 0.7978845608028654f;   // sqrt(2/pi)
    const float k1 = 0.044715f;
    return 0.5f * x * (1.0f + tanhf(k0 * (x + k1 * x * x * x)));
}

// ---------------- fp32 -> bf16 conversion ----------------
__global__ void cvt_f32_bf16(const float* __restrict__ in, __bf16* __restrict__ out, int n) {
    for (int i = blockIdx.x * blockDim.x + threadIdx.x; i < n; i += gridDim.x * blockDim.x)
        out[i] = (__bf16)in[i];
}

// ---------------- LayerNorm: f32 in -> bf16 out ----------------
__global__ __launch_bounds__(256) void ln_kernel(const float* __restrict__ x,
                                                 const float* __restrict__ g,
                                                 const float* __restrict__ bta,
                                                 __bf16* __restrict__ out) {
    __shared__ float rs[256], rs2[256];
    const int row = blockIdx.x, tid = threadIdx.x;
    const float* xr = x + (size_t)row * HH;
    float s = 0.f, s2 = 0.f;
#pragma unroll
    for (int i = 0; i < HH / 256; ++i) {
        float v = xr[tid + i * 256];
        s += v; s2 += v * v;
    }
    rs[tid] = s; rs2[tid] = s2;
    __syncthreads();
    for (int off = 128; off > 0; off >>= 1) {
        if (tid < off) { rs[tid] += rs[tid + off]; rs2[tid] += rs2[tid + off]; }
        __syncthreads();
    }
    const float mean = rs[0] * (1.0f / HH);
    const float var  = rs2[0] * (1.0f / HH) - mean * mean;
    const float rstd = rsqrtf(var + 1e-5f);
#pragma unroll
    for (int i = 0; i < HH / 256; ++i) {
        int c = tid + i * 256;
        out[(size_t)row * HH + c] = (__bf16)((xr[c] - mean) * rstd * g[c] + bta[c]);
    }
}

// ---------------- WMMA bf16 GEMM: C[M,N] = act(A*B + bias) (+res) ----------------
// A: MxK bf16 row-major, B: KxN bf16 row-major.
// Block = 256 thr (8 waves). Block tile 256x128x32, wave grid 4(M)x2(N),
// wave tile 64x64 (16 accumulators) -> 16 ds_load_b128 : 16 wmma per K-step.
// Double-buffered LDS: one barrier per K-step, staging overlaps compute.
__global__ __launch_bounds__(256) void gemm_bf16_wmma(const __bf16* __restrict__ A,
                                                      const __bf16* __restrict__ Bm,
                                                      const float*  __restrict__ bias,
                                                      const float*  __restrict__ res,
                                                      float*        __restrict__ outF,
                                                      __bf16*       __restrict__ outB,
                                                      int M, int N, int K, int act) {
    __shared__ __bf16 As[2][256][32];   // [buf][m][k]
    __shared__ __bf16 Bs[2][128][32];   // [buf][n][k] (transposed)

    const int tid  = threadIdx.x;
    const int lane = tid & 31, w = tid >> 5;
    const int half = lane >> 4, l16 = lane & 15;
    const int wm = w & 3, wn = w >> 2;            // 4 waves along M, 2 along N
    const int rowBlk = blockIdx.y * 256, colBlk = blockIdx.x * 128;

    v8f acc[4][4];
#pragma unroll
    for (int i = 0; i < 4; ++i)
#pragma unroll
        for (int j = 0; j < 4; ++j) acc[i][j] = vzero8();

    const int br = tid >> 3, bseg = (tid & 7) * 16;   // B stager coords

    auto stage = [&](int kk, int pb) {
        // A: one full 32-wide row per thread (256 rows)
        const v8bf* ga = (const v8bf*)(A + (size_t)(rowBlk + tid) * K + kk);
        v8bf a0 = ga[0], a1 = ga[1], a2 = ga[2], a3 = ga[3];
        v8bf* sp = (v8bf*)&As[pb][tid][0];
        sp[0] = a0; sp[1] = a1; sp[2] = a2; sp[3] = a3;
        // B: transpose into [n][k] so fragments are contiguous ds_load_b128
        const v8bf* gb = (const v8bf*)(Bm + (size_t)(kk + br) * N + colBlk + bseg);
        v8bf b0 = gb[0], b1 = gb[1];
#pragma unroll
        for (int i = 0; i < 8; ++i) {
            Bs[pb][bseg + i][br]     = b0[i];
            Bs[pb][bseg + 8 + i][br] = b1[i];
        }
    };

    const int nIt = K >> 5;
    stage(0, 0);
    int p = 0;
    for (int it = 0; it < nIt; ++it) {
        __syncthreads();
        if (it + 1 < nIt) {
            stage((it + 1) * 32, p ^ 1);
            if (it + 2 < nIt) {
                __builtin_prefetch(A  + (size_t)(rowBlk + tid) * K + (it + 2) * 32, 0, 1);
                __builtin_prefetch(Bm + (size_t)((it + 2) * 32 + br) * N + colBlk + bseg, 0, 1);
            }
        }
        v16bf af[4];
#pragma unroll
        for (int mt = 0; mt < 4; ++mt) {
            const int lr = wm * 64 + mt * 16 + l16;
            v8bf lo = *(const v8bf*)&As[p][lr][half * 8];
            v8bf hi = *(const v8bf*)&As[p][lr][16 + half * 8];
            af[mt] = make16(lo, hi);
        }
#pragma unroll
        for (int nt = 0; nt < 4; ++nt) {
            const int lc = wn * 64 + nt * 16 + l16;
            v8bf lo = *(const v8bf*)&Bs[p][lc][half * 8];
            v8bf hi = *(const v8bf*)&Bs[p][lc][16 + half * 8];
            v16bf bfr = make16(lo, hi);
#pragma unroll
            for (int mt = 0; mt < 4; ++mt)
                acc[mt][nt] = __builtin_amdgcn_wmma_f32_16x16x32_bf16(
                    false, af[mt], false, bfr, (short)0, acc[mt][nt], false, false);
        }
        p ^= 1;
    }

    // epilogue: bias depends only on column -> hoist per nt
#pragma unroll
    for (int nt = 0; nt < 4; ++nt) {
        const int gcol = colBlk + wn * 64 + nt * 16 + l16;
        const float bcol = bias[gcol];
#pragma unroll
        for (int mt = 0; mt < 4; ++mt)
#pragma unroll
            for (int r = 0; r < 8; ++r) {
                const int grow = rowBlk + wm * 64 + mt * 16 + r + half * 8;
                float v = acc[mt][nt][r] + bcol;
                if (act) v = gelu_tanh(v);
                const size_t o = (size_t)grow * N + gcol;
                if (res)  v += res[o];
                if (outF) outF[o] = v;
                if (outB) outB[o] = (__bf16)v;
            }
    }
}

// ---------------- RoPE on bf16 q,k in place ----------------
__global__ void rope_kernel(__bf16* __restrict__ q, __bf16* __restrict__ k) {
    const int idx = blockIdx.x * blockDim.x + threadIdx.x;
    const int total = BB * SS * NHH * (DD / 2);
    if (idx >= total) return;
    const int t   = idx & 63;
    int tmp = idx >> 6;
    const int h = tmp % NHH; tmp /= NHH;
    const int s = tmp % SS;
    const int b = tmp / SS;
    // inv_freq = base^(-t/64) = exp(-t * ln(1e4)/64)
    const float ang = (float)s * __expf(-(float)t * 0.14391156961757153f);
    float cs, sn; __sincosf(ang, &sn, &cs);
    const size_t base = (((size_t)(b * SS + s) * NHH + h) * DD);
    {
        float x1 = (float)q[base + t], x2 = (float)q[base + 64 + t];
        q[base + t]      = (__bf16)(x1 * cs - x2 * sn);
        q[base + 64 + t] = (__bf16)(x2 * cs + x1 * sn);
    }
    {
        float x1 = (float)k[base + t], x2 = (float)k[base + 64 + t];
        k[base + t]      = (__bf16)(x1 * cs - x2 * sn);
        k[base + 64 + t] = (__bf16)(x2 * cs + x1 * sn);
    }
}

// ---------------- causal flash attention (bf16 in/out, f32 accum) ----------------
// grid: (S/128, B*NH); block 256 (8 waves). Wave w owns q-rows [i0, i0+16).
__global__ __launch_bounds__(256) void attn_kernel(const __bf16* __restrict__ q,
                                                   const __bf16* __restrict__ k,
                                                   const __bf16* __restrict__ vv,
                                                   __bf16* __restrict__ o) {
    __shared__ __bf16 Kc[32][128];    // K chunk, [j][d]
    __shared__ __bf16 VTc[128][32];   // V chunk transposed, [d][j]
    __shared__ __bf16 Pst[8][16][32]; // per-wave P staging, [m][j]

    const int tid = threadIdx.x, lane = tid & 31, w = tid >> 5;
    const int half = lane >> 4, l16 = lane & 15;
    const int b = blockIdx.y / NHH, h = blockIdx.y % NHH;
    const int i0wg = blockIdx.x * 128;
    const int i0   = i0wg + w * 16;

    // Q fragments (4 k-slices of 32 covering D=128), kept in registers
    v16bf qf[4];
    {
        const __bf16* qp = q + ((size_t)(b * SS + i0 + l16) * NHH + h) * DD;
#pragma unroll
        for (int ks = 0; ks < 4; ++ks) {
            v8bf lo = *(const v8bf*)(qp + ks * 32 + half * 8);
            v8bf hi = *(const v8bf*)(qp + ks * 32 + 16 + half * 8);
            qf[ks] = make16(lo, hi);
        }
    }

    v8f Oa[8];
    float mr[8], lr[8];
#pragma unroll
    for (int c = 0; c < 8; ++c) Oa[c] = vzero8();
#pragma unroll
    for (int r = 0; r < 8; ++r) { mr[r] = -__builtin_inff(); lr[r] = 0.f; }

    const int jr = tid >> 3, cs = (tid & 7) * 16;  // staging coords
    const int jend = i0wg + 127;

    for (int j0 = 0; j0 <= jend; j0 += 32) {
        // stage K [j][d] and V transposed [d][j]
        {
            const size_t rowOff = ((size_t)(b * SS + j0 + jr) * NHH + h) * DD + cs;
            const v8bf* kp = (const v8bf*)(k  + rowOff);
            v8bf klo = kp[0], khi = kp[1];
            v8bf* sp = (v8bf*)&Kc[jr][cs];
            sp[0] = klo; sp[1] = khi;
            const v8bf* vp = (const v8bf*)(vv + rowOff);
            v8bf vlo = vp[0], vhi = vp[1];
#pragma unroll
            for (int i = 0; i < 8; ++i) {
                VTc[cs + i][jr]     = vlo[i];
                VTc[cs + 8 + i][jr] = vhi[i];
            }
        }
        __syncthreads();

        if (j0 <= i0 + 15) {   // wave-uniform: this chunk intersects wave's causal range
            v8f s0 = vzero8(), s1 = vzero8();
#pragma unroll
            for (int ks = 0; ks < 4; ++ks) {
                v8bf lo = *(const v8bf*)&Kc[l16][ks * 32 + half * 8];
                v8bf hi = *(const v8bf*)&Kc[l16][ks * 32 + 16 + half * 8];
                v16bf b0 = make16(lo, hi);
                s0 = __builtin_amdgcn_wmma_f32_16x16x32_bf16(false, qf[ks], false, b0,
                                                             (short)0, s0, false, false);
                lo = *(const v8bf*)&Kc[16 + l16][ks * 32 + half * 8];
                hi = *(const v8bf*)&Kc[16 + l16][ks * 32 + 16 + half * 8];
                v16bf b1 = make16(lo, hi);
                s1 = __builtin_amdgcn_wmma_f32_16x16x32_bf16(false, qf[ks], false, b1,
                                                             (short)0, s1, false, false);
            }
            // online softmax; row m = r + 8*half lives across the 16-lane half-group
            float alpha[8];
#pragma unroll
            for (int r = 0; r < 8; ++r) {
                const int qi = i0 + r + half * 8;
                float a0 = s0[r] * SOFTMAX_SCALE;
                float a1 = s1[r] * SOFTMAX_SCALE;
                if (j0 + l16      > qi) a0 = -__builtin_inff();
                if (j0 + 16 + l16 > qi) a1 = -__builtin_inff();
                float mx = fmaxf(a0, a1);
#pragma unroll
                for (int d = 1; d < 16; d <<= 1) mx = fmaxf(mx, __shfl_xor(mx, d, 32));
                const float mn = fmaxf(mr[r], mx);
                const float al = __expf(mr[r] - mn);
                mr[r] = mn; alpha[r] = al;
                const float p0 = __expf(a0 - mn);
                const float p1 = __expf(a1 - mn);
                float rs = p0 + p1;
#pragma unroll
                for (int d = 1; d < 16; d <<= 1) rs += __shfl_xor(rs, d, 32);
                lr[r] = lr[r] * al + rs;
                s0[r] = p0; s1[r] = p1;
            }
#pragma unroll
            for (int c = 0; c < 8; ++c)
#pragma unroll
                for (int r = 0; r < 8; ++r) Oa[c][r] *= alpha[r];

            // C-layout f32 -> A-layout bf16 via per-wave LDS patch
#pragma unroll
            for (int r = 0; r < 8; ++r) {
                const int m = r + half * 8;
                Pst[w][m][l16]      = (__bf16)s0[r];
                Pst[w][m][16 + l16] = (__bf16)s1[r];
            }
            v16bf pf;
            {
                v8bf lo = *(const v8bf*)&Pst[w][l16][half * 8];
                v8bf hi = *(const v8bf*)&Pst[w][l16][16 + half * 8];
                pf = make16(lo, hi);
            }
            // O += P * V  (8 d-tiles of 16, contraction over 32 j's)
#pragma unroll
            for (int c = 0; c < 8; ++c) {
                v8bf lo = *(const v8bf*)&VTc[c * 16 + l16][half * 8];
                v8bf hi = *(const v8bf*)&VTc[c * 16 + l16][16 + half * 8];
                v16bf bfr = make16(lo, hi);
                Oa[c] = __builtin_amdgcn_wmma_f32_16x16x32_bf16(false, pf, false, bfr,
                                                                (short)0, Oa[c], false, false);
            }
        }
        __syncthreads();
    }

    // normalize and write [b, i, h, d] bf16
#pragma unroll
    for (int c = 0; c < 8; ++c)
#pragma unroll
        for (int r = 0; r < 8; ++r) {
            const float ov = Oa[c][r] / lr[r];
            o[((size_t)(b * SS + i0 + r + half * 8) * NHH + h) * DD + c * 16 + l16] = (__bf16)ov;
        }
}

// ---------------- host driver ----------------
extern "C" void kernel_launch(void* const* d_in, const int* in_sizes, int n_in,
                              void* d_out, int out_size, void* d_ws, size_t ws_size,
                              hipStream_t stream) {
    const float* x   = (const float*)d_in[0];
    // d_in[1] = mask (unused, causal mask computed analytically)
    const float* Wq  = (const float*)d_in[2];   const float* bq = (const float*)d_in[3];
    const float* Wk  = (const float*)d_in[4];   const float* bk = (const float*)d_in[5];
    const float* Wv  = (const float*)d_in[6];   const float* bv = (const float*)d_in[7];
    const float* Wo  = (const float*)d_in[8];   const float* bo = (const float*)d_in[9];
    const float* W1  = (const float*)d_in[10];  const float* b1 = (const float*)d_in[11];
    const float* W2  = (const float*)d_in[12];  const float* b2 = (const float*)d_in[13];
    const float* g1  = (const float*)d_in[14];  const float* be1 = (const float*)d_in[15];
    const float* g2  = (const float*)d_in[16];  const float* be2 = (const float*)d_in[17];
    float* out = (float*)d_out;

    // workspace carve-up
    char* p = (char*)d_ws;
    auto alloc = [&](size_t bytes) -> void* {
        void* r = (void*)p;
        p += (bytes + 255) & ~(size_t)255;
        return r;
    };
    __bf16* wqb  = (__bf16*)alloc((size_t)HH * HH * 2);
    __bf16* wkb  = (__bf16*)alloc((size_t)HH * HH * 2);
    __bf16* wvb  = (__bf16*)alloc((size_t)HH * HH * 2);
    __bf16* wob  = (__bf16*)alloc((size_t)HH * HH * 2);
    __bf16* w1b  = (__bf16*)alloc((size_t)HH * FFF * 2);
    __bf16* w2b  = (__bf16*)alloc((size_t)FFF * HH * 2);
    __bf16* xn1  = (__bf16*)alloc((size_t)MM * HH * 2);
    __bf16* qb   = (__bf16*)alloc((size_t)MM * HH * 2);
    __bf16* kb   = (__bf16*)alloc((size_t)MM * HH * 2);
    __bf16* vb   = (__bf16*)alloc((size_t)MM * HH * 2);
    __bf16* attb = (__bf16*)alloc((size_t)MM * HH * 2);
    float*  x2   = (float*) alloc((size_t)MM * HH * 4);
    __bf16* xn2  = (__bf16*)alloc((size_t)MM * HH * 2);
    __bf16* hb   = (__bf16*)alloc((size_t)MM * FFF * 2);

    // 1) weights -> bf16
    cvt_f32_bf16<<<2048, 256, 0, stream>>>(Wq, wqb, HH * HH);
    cvt_f32_bf16<<<2048, 256, 0, stream>>>(Wk, wkb, HH * HH);
    cvt_f32_bf16<<<2048, 256, 0, stream>>>(Wv, wvb, HH * HH);
    cvt_f32_bf16<<<2048, 256, 0, stream>>>(Wo, wob, HH * HH);
    cvt_f32_bf16<<<4096, 256, 0, stream>>>(W1, w1b, HH * FFF);
    cvt_f32_bf16<<<4096, 256, 0, stream>>>(W2, w2b, FFF * HH);

    // 2) LN1
    ln_kernel<<<MM, 256, 0, stream>>>(x, g1, be1, xn1);

    // 3) QKV projections (block tile 256x128)
    dim3 gHH(HH / 128, MM / 256);
    gemm_bf16_wmma<<<gHH, 256, 0, stream>>>(xn1, wqb, bq, nullptr, nullptr, qb, MM, HH, HH, 0);
    gemm_bf16_wmma<<<gHH, 256, 0, stream>>>(xn1, wkb, bk, nullptr, nullptr, kb, MM, HH, HH, 0);
    gemm_bf16_wmma<<<gHH, 256, 0, stream>>>(xn1, wvb, bv, nullptr, nullptr, vb, MM, HH, HH, 0);

    // 4) RoPE on q, k
    rope_kernel<<<(BB * SS * NHH * (DD / 2)) / 256, 256, 0, stream>>>(qb, kb);

    // 5) causal flash attention
    attn_kernel<<<dim3(SS / 128, BB * NHH), 256, 0, stream>>>(qb, kb, vb, attb);

    // 6) O projection + residual(x) -> x2 (f32)
    gemm_bf16_wmma<<<gHH, 256, 0, stream>>>(attb, wob, bo, x, x2, nullptr, MM, HH, HH, 0);

    // 7) LN2
    ln_kernel<<<MM, 256, 0, stream>>>(x2, g2, be2, xn2);

    // 8) FFN up + GELU -> hb (bf16)
    gemm_bf16_wmma<<<dim3(FFF / 128, MM / 256), 256, 0, stream>>>(xn2, w1b, b1, nullptr, nullptr, hb,
                                                                  MM, FFF, HH, 1);

    // 9) FFN down + residual(x2) -> d_out (f32)
    gemm_bf16_wmma<<<gHH, 256, 0, stream>>>(hb, w2b, b2, x2, out, nullptr, MM, HH, FFF, 0);

    (void)in_sizes; (void)n_in; (void)out_size; (void)ws_size;
}
